// MultiHeadAttention_51264729645329
// MI455X (gfx1250) — compile-verified, matrix operand order checked by reading, not run on hardware
//
#include <hip/hip_runtime.h>
#include <hip/hip_bf16.h>

#define D_MODEL  1024
#define NUM_HEADS 16
#define D_K       64
#define SEQ     2048
#define BATCH      2
#define M_TOTAL (BATCH*SEQ)   // 4096

typedef __attribute__((ext_vector_type(16))) __bf16 v16bf;
typedef __attribute__((ext_vector_type(8)))  __bf16 v8bf;
typedef __attribute__((ext_vector_type(8)))  float  v8f;

// ---------- gfx1250 async global->LDS path (guarded; safe fallback) ----------
#if defined(__HIP_DEVICE_COMPILE__) && __has_builtin(__builtin_amdgcn_global_load_async_to_lds_b128)
#define HAVE_ASYNC_LDS 1
#else
#define HAVE_ASYNC_LDS 0
#endif

#if HAVE_ASYNC_LDS
typedef int v4i32 __attribute__((vector_size(16)));
typedef v4i32 __attribute__((address_space(1))) gv4i;   // global
typedef v4i32 __attribute__((address_space(3))) lv4i;   // LDS
#endif

// copy 16 bytes (8 bf16) global -> LDS, per-lane addresses
static __device__ inline void cp16(__bf16* l, const __bf16* g) {
#if HAVE_ASYNC_LDS
  __builtin_amdgcn_global_load_async_to_lds_b128((gv4i*)g, (lv4i*)l, 0, 0);
#else
  *reinterpret_cast<v8bf*>(l) = *reinterpret_cast<const v8bf*>(g);
#endif
}

static __device__ inline void async_fence() {
#if HAVE_ASYNC_LDS
#if __has_builtin(__builtin_amdgcn_s_wait_asynccnt)
  __builtin_amdgcn_s_wait_asynccnt(0);
#else
  asm volatile("s_wait_asynccnt 0" ::: "memory");
#endif
#endif
}

// ---------- WMMA wrapper: D = A(16x32 bf16) * B(32x16 bf16) + C(16x16 f32) ----------
static __device__ inline v8f wmma_bf16(v16bf a, v16bf b, v8f c) {
#if defined(__HIP_DEVICE_COMPILE__)
  return __builtin_amdgcn_wmma_f32_16x16x32_bf16(
      /*neg_a=*/false, a, /*neg_b=*/false, b,
      /*c_mod=*/(short)0, c, /*reuse_a=*/false, /*reuse_b=*/false);
#else
  return c;
#endif
}

// Fragment packing (ISA 16-bit A layout; B mirrored):
// lane = (row-or-col)%16, half = lane>>4
// elems 0..7  : K = half*8 + i ; elems 8..15 : K = 16 + half*8 + i
static __device__ inline v16bf frag_from_bf16(const __bf16* rowk0, int half) {
  v8bf lo = *reinterpret_cast<const v8bf*>(rowk0 + (half << 3));
  v8bf hi = *reinterpret_cast<const v8bf*>(rowk0 + 16 + (half << 3));
  return __builtin_shufflevector(lo, hi, 0,1,2,3,4,5,6,7,8,9,10,11,12,13,14,15);
}

// ============================================================================
// Kernel 0: bulk f32 -> bf16 conversion, 8 elems/thread
// ============================================================================
__global__ __launch_bounds__(256) void cvt_bf16_kernel(
    const float* __restrict__ src, __bf16* __restrict__ dst, int n)
{
  const int i = (blockIdx.x * 256 + threadIdx.x) * 8;
  if (i >= n) return;
  const float4 a = *reinterpret_cast<const float4*>(src + i);
  const float4 b = *reinterpret_cast<const float4*>(src + i + 4);
  v8bf o;
  o[0]=(__bf16)a.x; o[1]=(__bf16)a.y; o[2]=(__bf16)a.z; o[3]=(__bf16)a.w;
  o[4]=(__bf16)b.x; o[5]=(__bf16)b.y; o[6]=(__bf16)b.z; o[7]=(__bf16)b.w;
  *reinterpret_cast<v8bf*>(dst + i) = o;
}

// ============================================================================
// Kernel 1: Y = X @ W^T + bias. W tile staged in LDS (shared by 4 waves).
// Wave tile 32(M) x 64(N); block = 4 wave32 -> 128 x 64 block tile.
// mode 0: bf16 head layout [B,H,S,Dk] | mode 1: bf16 [B,H,Dk,S] | mode 2: f32 [M,D]
// ============================================================================
__global__ __launch_bounds__(128) void gemm_kernel(
    const __bf16* __restrict__ A, const __bf16* __restrict__ W,
    const float* __restrict__ bias, __bf16* __restrict__ dstb,
    float* __restrict__ dstf, int mode)
{
  __shared__ __bf16 wbuf[64 * 32];     // 64 N-rows x 32 K  (4 KB)

  const int lane = threadIdx.x & 31;
  const int wave = threadIdx.x >> 5;
  const int half = lane >> 4;
  const int l16  = lane & 15;
  const int t    = threadIdx.x;
  const int n0 = blockIdx.x * 64;
  const int m0 = blockIdx.y * 128 + wave * 32;

  v8f acc[2][4];
  const v8f vzero = {0.f,0.f,0.f,0.f,0.f,0.f,0.f,0.f};
#pragma unroll
  for (int i = 0; i < 2; ++i)
#pragma unroll
    for (int j = 0; j < 4; ++j) acc[i][j] = vzero;

  const __bf16* arow0 = A + (size_t)(m0 + l16) * D_MODEL;
  const __bf16* arow1 = A + (size_t)(m0 + 16 + l16) * D_MODEL;
  const __bf16* wbase = W + (size_t)n0 * D_MODEL;

  for (int k0 = 0; k0 < D_MODEL; k0 += 32) {
    __syncthreads();                   // previous iteration done reading LDS
    // stage 64x32 W tile: 256 16B chunks, 2 per thread
#pragma unroll
    for (int c = 0; c < 2; ++c) {
      const int chunk = t * 2 + c;           // 0..255
      const int row   = chunk >> 2;          // 64 rows, 4 chunks each
      const int col   = (chunk & 3) * 8;     // element offset in row
      cp16(&wbuf[row * 32 + col], wbase + (size_t)row * D_MODEL + k0 + col);
    }
    async_fence();
    __syncthreads();

    const v16bf a0 = frag_from_bf16(arow0 + k0, half);
    const v16bf a1 = frag_from_bf16(arow1 + k0, half);
#pragma unroll
    for (int nt = 0; nt < 4; ++nt) {
      const v16bf b = frag_from_bf16(&wbuf[(nt * 16 + l16) * 32], half);
      acc[0][nt] = wmma_bf16(a0, b, acc[0][nt]);
      acc[1][nt] = wmma_bf16(a1, b, acc[1][nt]);
    }
  }

#pragma unroll
  for (int nt = 0; nt < 4; ++nt) {
    const int n  = n0 + nt * 16 + l16;
    const float bv = bias[n];
    const int h  = n >> 6;
    const int dk = n & 63;
#pragma unroll
    for (int mt = 0; mt < 2; ++mt) {
#pragma unroll
      for (int r = 0; r < 8; ++r) {
        const int m  = m0 + mt * 16 + r + 8 * half;   // C layout: M = r + 8*half
        const float y = acc[mt][nt][r] + bv;
        if (mode == 0) {
          const int b_ = m >> 11, s = m & (SEQ - 1);
          dstb[(((size_t)b_ * NUM_HEADS + h) * SEQ + s) * D_K + dk] = (__bf16)y;
        } else if (mode == 1) {
          const int b_ = m >> 11, s = m & (SEQ - 1);
          dstb[(((size_t)b_ * NUM_HEADS + h) * D_K + dk) * SEQ + s] = (__bf16)y;
        } else {
          dstf[(size_t)m * D_MODEL + n] = y;
        }
      }
    }
  }
}

// ============================================================================
// Kernel 2: flash attention per (b,h). 4 waves x 16 queries; K/V tiles staged
// in LDS and shared by all 4 waves (4x cut in L2 traffic).
//   S^T = K_tile * Q^T    (C layout: lane=query, VGPR=key)
//   online softmax over keys, exp2 domain (1 shfl_xor(16) per reduction)
//   O^T += V^T_tile * P^T (P^T C-tile packing == B-fragment packing)
// ============================================================================
__global__ __launch_bounds__(128) void attn_kernel(
    const __bf16* __restrict__ Qh, const __bf16* __restrict__ Kh,
    const __bf16* __restrict__ Vt, __bf16* __restrict__ AO)
{
  __shared__ __bf16 kbuf[32 * 64];     // 32 keys x 64 dk   (4 KB)
  __shared__ __bf16 vbuf[64 * 32];     // 64 d   x 32 keys  (4 KB)

  const int lane = threadIdx.x & 31;
  const int wave = threadIdx.x >> 5;
  const int half = lane >> 4;
  const int l16  = lane & 15;
  const int t    = threadIdx.x;
  const int bh = blockIdx.y;           // 0..31
  const int b_ = bh >> 4, h = bh & 15;
  const int q0 = (blockIdx.x * 4 + wave) * 16;

  // Q^T as B fragments (WMMA K-dim = d_k)
  const __bf16* qptr = Qh + ((size_t)bh * SEQ + q0 + l16) * D_K;
  const v16bf qb0 = frag_from_bf16(qptr,      half);   // dk 0..31
  const v16bf qb1 = frag_from_bf16(qptr + 32, half);   // dk 32..63

  v8f o[4];
  const v8f vzero = {0.f,0.f,0.f,0.f,0.f,0.f,0.f,0.f};
#pragma unroll
  for (int i = 0; i < 4; ++i) o[i] = vzero;

  float m_run = -1e30f;                 // running max, exp2 domain
  float l_run = 0.0f;
  const float cl2 = 0.125f * 1.44269504088896340736f;  // scale * log2(e)

  const __bf16* kglob = Kh + (size_t)bh * SEQ * D_K;   // [key][dk]
  const __bf16* vglob = Vt + (size_t)bh * D_K * SEQ;   // [d][s]

  for (int j = 0; j < SEQ; j += 32) {
    __syncthreads();                   // previous iteration done reading LDS
    // stage K tile (32x64) + V^T tile (64x32): 2 chunks each per thread
#pragma unroll
    for (int c = 0; c < 2; ++c) {
      const int kc = t * 2 + c;              // 0..255
      const int krow = kc >> 3, kcol = (kc & 7) * 8;
      cp16(&kbuf[krow * 64 + kcol], kglob + (size_t)(j + krow) * D_K + kcol);
      const int vrow = kc >> 2, vcol = (kc & 3) * 8;
      cp16(&vbuf[vrow * 32 + vcol], vglob + (size_t)vrow * SEQ + j + vcol);
    }
    // pull next K tile toward the WGP while this one is consumed
    if (j + 32 < SEQ)
      __builtin_prefetch(kglob + (size_t)(j + 32) * D_K + t * 16, 0, 1);
    async_fence();
    __syncthreads();

    // ---- scores for keys [j, j+32): two 16x16 tiles (A frags from LDS) --
    v16bf ka0 = frag_from_bf16(&kbuf[l16 * 64],            half);
    v16bf ka1 = frag_from_bf16(&kbuf[l16 * 64 + 32],       half);
    v16bf kc0 = frag_from_bf16(&kbuf[(16 + l16) * 64],      half);
    v16bf kc1 = frag_from_bf16(&kbuf[(16 + l16) * 64 + 32], half);
    v8f s0 = vzero, s1 = vzero;
    s0 = wmma_bf16(ka0, qb0, s0);
    s0 = wmma_bf16(ka1, qb1, s0);
    s1 = wmma_bf16(kc0, qb0, s1);
    s1 = wmma_bf16(kc1, qb1, s1);

    // ---- online softmax over the 32 keys (exp2 domain) -----------------
    float t0[8], t1[8];
    float m_blk = -1e30f;
#pragma unroll
    for (int r = 0; r < 8; ++r) {
      t0[r] = s0[r] * cl2;  m_blk = fmaxf(m_blk, t0[r]);
      t1[r] = s1[r] * cl2;  m_blk = fmaxf(m_blk, t1[r]);
    }
    m_blk = fmaxf(m_blk, __shfl_xor(m_blk, 16, 32));   // combine key halves
    const float m_new = fmaxf(m_run, m_blk);
    const float alpha = exp2f(m_run - m_new);

    float p0[8], p1[8], lsum = 0.f;
#pragma unroll
    for (int r = 0; r < 8; ++r) {
      p0[r] = exp2f(t0[r] - m_new);  lsum += p0[r];
      p1[r] = exp2f(t1[r] - m_new);  lsum += p1[r];
    }
    lsum += __shfl_xor(lsum, 16, 32);
    l_run = l_run * alpha + lsum;
    m_run = m_new;

#pragma unroll
    for (int f = 0; f < 4; ++f)
#pragma unroll
      for (int r = 0; r < 8; ++r) o[f][r] *= alpha;

    // ---- P^T as B fragment: C-tile packing matches B packing -----------
    v16bf pb;
#pragma unroll
    for (int i = 0; i < 8; ++i) { pb[i] = (__bf16)p0[i]; pb[8 + i] = (__bf16)p1[i]; }

    // ---- O^T += V^T * P^T (4 d_k tiles of 16, A frags from LDS) ---------
#pragma unroll
    for (int dt = 0; dt < 4; ++dt) {
      v16bf va = frag_from_bf16(&vbuf[(dt * 16 + l16) * 32], half);
      o[dt] = wmma_bf16(va, pb, o[dt]);
    }
  }

  // ---- normalize + store merged-head layout [B,S,D] (bf16) --------------
  const float inv_l = 1.0f / l_run;
  const size_t row = ((size_t)b_ * SEQ + q0 + l16) * D_MODEL + (size_t)h * D_K;
#pragma unroll
  for (int dt = 0; dt < 4; ++dt) {
#pragma unroll
    for (int r = 0; r < 8; ++r) {
      const int d = dt * 16 + r + 8 * half;
      AO[row + d] = (__bf16)(o[dt][r] * inv_l);
    }
  }
}

// ============================================================================
extern "C" void kernel_launch(void* const* d_in, const int* in_sizes, int n_in,
                              void* d_out, int out_size, void* d_ws, size_t ws_size,
                              hipStream_t stream) {
  (void)in_sizes; (void)n_in; (void)out_size; (void)ws_size;
  const float* q  = (const float*)d_in[0];
  const float* k  = (const float*)d_in[1];
  const float* v  = (const float*)d_in[2];
  const float* Wq = (const float*)d_in[3];
  const float* bq = (const float*)d_in[4];
  const float* Wk = (const float*)d_in[5];
  const float* bk = (const float*)d_in[6];
  const float* Wv = (const float*)d_in[7];
  const float* bv = (const float*)d_in[8];
  const float* Wo = (const float*)d_in[9];
  const float* bo = (const float*)d_in[10];
  float* out = (float*)d_out;

  char* ws = (char*)d_ws;
  const size_t actN = (size_t)M_TOTAL * D_MODEL;          // 4M elems
  const size_t wN   = (size_t)D_MODEL * D_MODEL;          // 1M elems
  const size_t actB = actN * sizeof(__bf16);              // 8 MB
  const size_t wB   = wN * sizeof(__bf16);                // 2 MB

  __bf16* Xq  = (__bf16*)(ws);
  __bf16* Xk  = (__bf16*)(ws + actB);
  __bf16* Xv  = (__bf16*)(ws + 2*actB);
  __bf16* Wqb = (__bf16*)(ws + 3*actB);
  __bf16* Wkb = (__bf16*)(ws + 3*actB + wB);
  __bf16* Wvb = (__bf16*)(ws + 3*actB + 2*wB);
  __bf16* Wob = (__bf16*)(ws + 3*actB + 3*wB);
  __bf16* Qh  = (__bf16*)(ws + 3*actB + 4*wB);
  __bf16* Kh  = (__bf16*)(ws + 4*actB + 4*wB);
  __bf16* Vt  = (__bf16*)(ws + 5*actB + 4*wB);
  __bf16* AO  = (__bf16*)(ws + 6*actB + 4*wB);            // total 64 MB

  // ---- one-time f32 -> bf16 conversion ----------------------------------
  const int actBlocks = (int)(actN / 8 / 256);
  const int wBlocks   = (int)(wN / 8 / 256);
  cvt_bf16_kernel<<<actBlocks, 256, 0, stream>>>(q,  Xq,  (int)actN);
  cvt_bf16_kernel<<<actBlocks, 256, 0, stream>>>(k,  Xk,  (int)actN);
  cvt_bf16_kernel<<<actBlocks, 256, 0, stream>>>(v,  Xv,  (int)actN);
  cvt_bf16_kernel<<<wBlocks,   256, 0, stream>>>(Wq, Wqb, (int)wN);
  cvt_bf16_kernel<<<wBlocks,   256, 0, stream>>>(Wk, Wkb, (int)wN);
  cvt_bf16_kernel<<<wBlocks,   256, 0, stream>>>(Wv, Wvb, (int)wN);
  cvt_bf16_kernel<<<wBlocks,   256, 0, stream>>>(Wo, Wob, (int)wN);

  // ---- projections -> attention -> output projection --------------------
  const dim3 gproj(D_MODEL / 64, M_TOTAL / 128);          // 16 x 32 blocks
  gemm_kernel<<<gproj, 128, 0, stream>>>(Xq, Wqb, bq, Qh, nullptr, 0);
  gemm_kernel<<<gproj, 128, 0, stream>>>(Xk, Wkb, bk, Kh, nullptr, 0);
  gemm_kernel<<<gproj, 128, 0, stream>>>(Xv, Wvb, bv, Vt, nullptr, 1);
  attn_kernel<<<dim3(SEQ / 64, BATCH * NUM_HEADS), 128, 0, stream>>>(Qh, Kh, Vt, AO);
  gemm_kernel<<<gproj, 128, 0, stream>>>(AO, Wob, bo, nullptr, out, 2);
}
